// XFeat_26139170964070
// MI455X (gfx1250) — matrix operand order repaired and checked by reading, not compiled, not run on hardware
//
#include <hip/hip_runtime.h>
#include <math.h>

#define BATCH 8
#define CH 64
#define HC 96
#define WC 96
#define HH 768
#define WW 768
#define NPIX (HH*WW)          // 589824
#define TOPK 2048
#define EPSN 1e-12f
#define THRESHV 0.05f

typedef __attribute__((ext_vector_type(16))) _Float16 v16h;
typedef __attribute__((ext_vector_type(8)))  float    v8f;

// ---- float <-> order-preserving uint key (descending = larger key) ----
__device__ __forceinline__ unsigned f2k(float f){
  unsigned u = __float_as_uint(f);
  return (u & 0x80000000u) ? ~u : (u | 0x80000000u);
}
__device__ __forceinline__ float k2f(unsigned k){
  return (k & 0x80000000u) ? __uint_as_float(k & 0x7FFFFFFFu) : __uint_as_float(~k);
}

// ---- cubic kernel weights, A = -0.75 (matches reference _cubic_w) ----
__device__ __forceinline__ void cubic_w(float t, float* w){
  const float A_ = -0.75f;
  auto k1 = [&](float s){ return ((A_ + 2.f) * s - (A_ + 3.f)) * s * s + 1.f; };
  auto k2 = [&](float s){ return ((A_ * s - 5.f * A_) * s + 8.f * A_) * s - 4.f * A_; };
  w[0] = k2(t + 1.f); w[1] = k1(t); w[2] = k1(1.f - t); w[3] = k2(2.f - t);
}

// ---------------- K0: M1n = M1 / max(||M1||_c, eps) ----------------
__global__ void k_norm_m1(const float* __restrict__ M1, float* __restrict__ M1n){
  int i = blockIdx.x * blockDim.x + threadIdx.x;      // b*9216 + yc*96 + xc
  if (i >= BATCH * HC * WC) return;
  int b = i / (HC * WC); int p = i % (HC * WC);
  const float* src = M1 + (size_t)b * CH * HC * WC + p;
  float ss = 0.f;
  #pragma unroll
  for (int c = 0; c < CH; c++){ float v = src[(size_t)c * HC * WC]; ss += v * v; }
  float inv = 1.f / fmaxf(sqrtf(ss), EPSN);
  float* dst = M1n + (size_t)b * CH * HC * WC + p;
  #pragma unroll
  for (int c = 0; c < CH; c++) dst[(size_t)c * HC * WC] = src[(size_t)c * HC * WC] * inv;
}

// ---------------- K1: softmax(K1,65)[:64] -> 8x8 pixel-shuffle heatmap ----------------
__global__ void k_heatmap(const float* __restrict__ K1, float* __restrict__ hm){
  int i = blockIdx.x * blockDim.x + threadIdx.x;      // cell index
  if (i >= BATCH * HC * WC) return;
  int b = i / (HC * WC); int p = i % (HC * WC);
  int yc = p / WC, xc = p % WC;
  const float* src = K1 + (size_t)b * 65 * HC * WC + p;
  float mx = -INFINITY;
  for (int c = 0; c < 65; c++) mx = fmaxf(mx, src[(size_t)c * HC * WC]);
  float s = 0.f;
  for (int c = 0; c < 65; c++) s += expf(src[(size_t)c * HC * WC] - mx);
  float inv = 1.f / s;
  for (int k = 0; k < 64; k++){
    int dy = k >> 3, dx = k & 7;
    hm[(size_t)b * NPIX + (size_t)(yc * 8 + dy) * WW + (xc * 8 + dx)] =
        expf(src[(size_t)k * HC * WC] - mx) * inv;
  }
}

// ---------------- K2: 5x5 NMS + nearest(hm) * bilinear(H1) scoring ----------------
__global__ void k_scores(const float* __restrict__ hm, const float* __restrict__ H1,
                         float* __restrict__ scores){
  int i = blockIdx.x * blockDim.x + threadIdx.x;
  if (i >= BATCH * NPIX) return;
  int b = i / NPIX; int p = i % NPIX; int y = p / WW, x = p % WW;
  const float* hb = hm + (size_t)b * NPIX;
  float v = hb[p];
  float lm = -INFINITY;
  for (int dy = -2; dy <= 2; dy++){
    int yy = y + dy; if ((unsigned)yy >= HH) continue;
    for (int dx = -2; dx <= 2; dx++){
      int xx = x + dx; if ((unsigned)xx >= WW) continue;
      lm = fmaxf(lm, hb[yy * WW + xx]);
    }
  }
  bool posf = (v == lm) && (v > THRESHV);
  float sc = -1.f;
  if (posf){
    // nearest sample of hm at full res: fx = x*W/(W-1) - 0.5, round half-even
    float fx = (float)x * (float)WW / (float)(WW - 1) - 0.5f;
    float fy = (float)y * (float)HH / (float)(HH - 1) - 0.5f;
    int ix = (int)rintf(fx), iy = (int)rintf(fy);
    float sn = ((unsigned)ix < WW && (unsigned)iy < HH) ? hb[iy * WW + ix] : 0.f;
    // bilinear sample of H1 (96x96), zero out-of-bounds taps
    float gx = (float)x * (float)WC / (float)(WW - 1) - 0.5f;
    float gy = (float)y * (float)HC / (float)(HH - 1) - 0.5f;
    float x0f = floorf(gx), y0f = floorf(gy);
    float tx = gx - x0f, ty = gy - y0f;
    int x0 = (int)x0f, y0 = (int)y0f;
    const float* Hb = H1 + (size_t)b * HC * WC;
    auto g2 = [&](int yy, int xx)->float{
      return ((unsigned)yy < HC && (unsigned)xx < WC) ? Hb[yy * WC + xx] : 0.f;
    };
    float sb = (g2(y0, x0) * (1.f - tx) + g2(y0, x0 + 1) * tx) * (1.f - ty)
             + (g2(y0 + 1, x0) * (1.f - tx) + g2(y0 + 1, x0 + 1) * tx) * ty;
    sc = sn * sb;
  }
  if (p == 0) sc = -1.f;
  scores[i] = sc;
}

// ---------------- K3: exact radix-select top-2048 per batch (deterministic) ----------------
__global__ __launch_bounds__(1024) void k_select(const float* __restrict__ scores,
                                                 unsigned* __restrict__ selKey,
                                                 unsigned* __restrict__ selIdx){
  int b = blockIdx.x;
  const float* sb = scores + (size_t)b * NPIX;
  __shared__ unsigned hist[256];
  __shared__ unsigned sh_prefix, sh_quota;
  __shared__ unsigned scanG[1024], scanE[1024];
  __shared__ unsigned baseG, baseE;
  int tid = threadIdx.x;
  if (tid == 0){ sh_prefix = 0u; sh_quota = TOPK; }
  __syncthreads();
  // 4 passes of 8-bit radix refinement, MSB first
  for (int pass = 0; pass < 4; pass++){
    int shift = 24 - 8 * pass;
    if (tid < 256) hist[tid] = 0u;
    __syncthreads();
    unsigned pref = sh_prefix;
    for (int i = tid; i < NPIX; i += 1024){
      unsigned key = f2k(sb[i]);
      bool m = true;
      if (pass) m = ((key >> (shift + 8)) == pref);
      if (m) atomicAdd(&hist[(key >> shift) & 255u], 1u);
    }
    __syncthreads();
    if (tid == 0){
      unsigned q = sh_quota, cum = 0u;
      for (int d = 255; d >= 0; d--){
        unsigned h = hist[d];
        if (cum + h >= q){ sh_prefix = (pref << 8) | (unsigned)d; sh_quota = q - cum; break; }
        cum += h;
      }
    }
    __syncthreads();
  }
  unsigned thresh = sh_prefix;
  unsigned neq = sh_quota;              // # elements == thresh to take (lowest indices)
  unsigned ng  = TOPK - neq;            // # elements strictly greater
  if (tid == 0){ baseG = 0u; baseE = 0u; }
  __syncthreads();
  // stable index-ordered compaction via block scans
  for (int start = 0; start < NPIX; start += 1024){
    int i = start + tid;
    unsigned key = 0u; bool inb = (i < NPIX);
    if (inb) key = f2k(sb[i]);
    unsigned g = (inb && key >  thresh) ? 1u : 0u;
    unsigned e = (inb && key == thresh) ? 1u : 0u;
    scanG[tid] = g; scanE[tid] = e;
    __syncthreads();
    for (int off = 1; off < 1024; off <<= 1){
      unsigned vg = (tid >= off) ? scanG[tid - off] : 0u;
      unsigned ve = (tid >= off) ? scanE[tid - off] : 0u;
      __syncthreads();
      scanG[tid] += vg; scanE[tid] += ve;
      __syncthreads();
    }
    if (g){
      unsigned p = baseG + scanG[tid] - 1u;
      selKey[b * TOPK + p] = key; selIdx[b * TOPK + p] = (unsigned)i;
    }
    if (e){
      unsigned q = baseE + scanE[tid] - 1u;
      if (q < neq){
        unsigned p = ng + q;
        selKey[b * TOPK + p] = key; selIdx[b * TOPK + p] = (unsigned)i;
      }
    }
    __syncthreads();
    if (tid == 0){ baseG += scanG[1023]; baseE += scanE[1023]; }
    __syncthreads();
  }
}

// ---------------- K4: bitonic sort 2048 (key desc, idx asc) per batch in LDS ----------------
__global__ __launch_bounds__(1024) void k_sort(unsigned* __restrict__ selKey,
                                               unsigned* __restrict__ selIdx){
  int b = blockIdx.x; int tid = threadIdx.x;
  __shared__ unsigned long long s[TOPK];   // 16 KB of the 320 KB WGP LDS
  for (int t = tid; t < TOPK; t += 1024){
    unsigned k  = selKey[b * TOPK + t];
    unsigned id = selIdx[b * TOPK + t];
    s[t] = ((unsigned long long)k << 32) | (unsigned long long)(0xFFFFFFFFu - id);
  }
  __syncthreads();
  for (int k = 2; k <= TOPK; k <<= 1){
    for (int j = k >> 1; j > 0; j >>= 1){
      for (int t = tid; t < TOPK; t += 1024){
        int partner = t ^ j;
        if (partner > t){
          bool descBlock = ((t & k) == 0);
          unsigned long long a = s[t], c = s[partner];
          bool doSwap = descBlock ? (a < c) : (a > c);
          if (doSwap){ s[t] = c; s[partner] = a; }
        }
      }
      __syncthreads();
    }
  }
  for (int t = tid; t < TOPK; t += 1024){
    unsigned long long v = s[t];
    selKey[b * TOPK + t] = (unsigned)(v >> 32);
    selIdx[b * TOPK + t] = 0xFFFFFFFFu - (unsigned)(v & 0xFFFFFFFFull);
  }
}

// ---------------- K5: emit mkpts / sc_top / valid ----------------
__global__ void k_emit(const unsigned* __restrict__ selKey, const unsigned* __restrict__ selIdx,
                       float* __restrict__ out){
  int i = blockIdx.x * blockDim.x + threadIdx.x;   // b*TOPK + t
  if (i >= BATCH * TOPK) return;
  unsigned key = selKey[i]; unsigned idx = selIdx[i];
  float sc = k2f(key);
  float x = (float)(idx % WW), y = (float)(idx / WW);
  float* mk = out;                                             // (B,TOPK,2)
  float* st = out + (size_t)BATCH * TOPK * 2;                  // (B,TOPK)
  float* va = out + (size_t)BATCH * TOPK * 2 + (size_t)BATCH * TOPK
                  + (size_t)BATCH * TOPK * 64;                 // (B,TOPK)
  mk[2 * i] = x; mk[2 * i + 1] = y;
  st[i] = sc;
  va[i] = (sc > 0.f) ? 1.f : 0.f;
}

// ---------------- K6: bicubic descriptor interpolation via block-diagonal WMMA ----------------
// Per wave: D(16 kp x 16 ch) = sum_{kk=0..7} A_kk(16x32) * B_kk(32x16), K=256 = 16 kp * 16 taps.
// A nonzero only at kk == m>>1 per lane -> built once, selected by v_cndmask.
// B gathers are 3-phase (addresses -> batched loads -> convert) so all 16 loads
// stay in flight before a single staged s_wait_loadcnt, hiding L2 latency.
__global__ __launch_bounds__(128) void k_feats(const float* __restrict__ M1n,
                                               const unsigned* __restrict__ selIdx,
                                               float* __restrict__ outFeats){
  int g = blockIdx.x;            // keypoint group (16 kpts)
  int b = blockIdx.y;            // batch
  int tid  = threadIdx.x;
  int lane = tid & 31;
  int wave = tid >> 5;           // 4 waves -> channel groups 0,16,32,48
  __shared__ int   sx0[16], sy0[16];
  __shared__ __align__(16) float sw[16][16];   // [kp][tap], tap = j*4+i, weight wy[j]*wx[i]
  __shared__ float sfeat[16][64];
  __shared__ float snorm[16];

  if (tid < 16){
    unsigned idx = selIdx[(size_t)b * TOPK + g * 16 + tid];
    float x = (float)(idx % WW), y = (float)(idx / WW);
    float fx = x * (float)WC / (float)(WW - 1) - 0.5f;   // sample into 96x96 grid
    float fy = y * (float)HC / (float)(HH - 1) - 0.5f;
    float x0f = floorf(fx), y0f = floorf(fy);
    float tx = fx - x0f, ty = fy - y0f;
    sx0[tid] = (int)x0f; sy0[tid] = (int)y0f;
    float wx[4], wy[4];
    cubic_w(tx, wx); cubic_w(ty, wy);
    #pragma unroll
    for (int j = 0; j < 4; j++)
      #pragma unroll
      for (int i2 = 0; i2 < 4; i2++)
        sw[tid][j * 4 + i2] = wy[j] * wx[i2];
  }
  __syncthreads();

  int cbase = wave * 16;
  int m   = lane & 15;           // A row / D column index
  int hiK = lane >> 4;           // half-wave selector
  int cch = cbase + m;           // this lane's channel for B and C/D
  const float* Mb = M1n + (size_t)b * CH * HC * WC;
  const float* plane = Mb + (size_t)cch * (HC * WC);
  __builtin_prefetch(plane, 0, 0);             // global_prefetch_b8

  // Load this lane's weight row from LDS once (4x ds_load_b128) into registers.
  float wrow[16];
  #pragma unroll
  for (int q4 = 0; q4 < 4; q4++){
    float4 q = ((const float4*)(&sw[m][0]))[q4];
    wrow[q4 * 4 + 0] = q.x; wrow[q4 * 4 + 1] = q.y;
    wrow[q4 * 4 + 2] = q.z; wrow[q4 * 4 + 3] = q.w;
  }
  // Build the single nonzero A tile for this lane (active only when kk == m>>1).
  v16h aAct;
  #pragma unroll
  for (int h = 0; h < 16; h++){
    int K = h + (hiK << 4);                              // 0..31 contiguous, matches B
    aAct[h] = (_Float16)(((K >> 4) == (m & 1)) ? wrow[K & 15] : 0.f);
  }
  v16h zerov = {};
  int myKK = m >> 1;

  v8f acc = {};
  #pragma unroll
  for (int kk = 0; kk < 8; kk++){
    v16h a = (kk == myKK) ? aAct : zerov;                // 8x v_cndmask, no LDS, no branch
    // B tile (32x16 f16): lane L column N=L%16; element h -> K = h + 16*hiK
    int kp = kk * 2 + hiK;
    int x0 = sx0[kp], y0 = sy0[kp];

    // Phase 1: addresses + masks (pure VALU)
    int  offs[16];
    bool inb[16];
    #pragma unroll
    for (int h = 0; h < 16; h++){
      int j = h >> 2, i2 = h & 3;                        // tap = j*4+i
      int yy = y0 + j - 1, xx = x0 + i2 - 1;
      inb[h] = ((unsigned)yy < HC) & ((unsigned)xx < WC);
      int ycl = min(max(yy, 0), HC - 1);
      int xcl = min(max(xx, 0), WC - 1);
      offs[h] = ycl * WC + xcl;
    }
    // Phase 2: 16 independent loads in flight (batched vmem)
    float vals[16];
    #pragma unroll
    for (int h = 0; h < 16; h++) vals[h] = plane[offs[h]];
    // Phase 3: mask + convert into B operand
    v16h bm;
    #pragma unroll
    for (int h = 0; h < 16; h++)
      bm[h] = (_Float16)(inb[h] ? vals[h] : 0.f);

    acc = __builtin_amdgcn_wmma_f32_16x16x32_f16(false, a, false, bm,
                                                 (short)0, acc, false, false);
  }

  // D layout: VGPR r on lane L -> (M = r + 8*(L>=16), N = L%16)
  #pragma unroll
  for (int r = 0; r < 8; r++){
    int kp = r + (hiK << 3);
    sfeat[kp][cch] = acc[r];
  }
  __syncthreads();
  if (tid < 16){
    float ss = 0.f;
    #pragma unroll
    for (int c = 0; c < 64; c++){ float v = sfeat[tid][c]; ss += v * v; }
    snorm[tid] = fmaxf(sqrtf(ss), EPSN);
  }
  __syncthreads();
  for (int e = tid; e < 16 * 64; e += 128){
    int kp = e >> 6, c = e & 63;
    outFeats[((size_t)b * TOPK + g * 16 + kp) * 64 + c] = sfeat[kp][c] / snorm[kp];
  }
}

extern "C" void kernel_launch(void* const* d_in, const int* in_sizes, int n_in,
                              void* d_out, int out_size, void* d_ws, size_t ws_size,
                              hipStream_t stream){
  const float* M1 = (const float*)d_in[0];   // (8,64,96,96)
  const float* K1 = (const float*)d_in[1];   // (8,65,96,96)
  const float* H1 = (const float*)d_in[2];   // (8,1,96,96)
  float* out = (float*)d_out;                // mkpts | sc_top | feats | valid

  float* wsf    = (float*)d_ws;
  float* M1n    = wsf;                                          // 8*64*96*96
  float* hm     = M1n + (size_t)BATCH * CH * HC * WC;           // 8*768*768
  float* scores = hm  + (size_t)BATCH * NPIX;                   // 8*768*768
  unsigned* selKey = (unsigned*)(scores + (size_t)BATCH * NPIX);
  unsigned* selIdx = selKey + BATCH * TOPK;

  int ncell = BATCH * HC * WC;
  k_norm_m1<<<(ncell + 255) / 256, 256, 0, stream>>>(M1, M1n);
  k_heatmap<<<(ncell + 255) / 256, 256, 0, stream>>>(K1, hm);
  int npx = BATCH * NPIX;
  k_scores<<<(npx + 255) / 256, 256, 0, stream>>>(hm, H1, scores);
  k_select<<<BATCH, 1024, 0, stream>>>(scores, selKey, selIdx);
  k_sort<<<BATCH, 1024, 0, stream>>>(selKey, selIdx);
  k_emit<<<(BATCH * TOPK + 255) / 256, 256, 0, stream>>>(selKey, selIdx, out);
  float* outFeats = out + (size_t)BATCH * TOPK * 2 + (size_t)BATCH * TOPK;
  dim3 grid(TOPK / 16, BATCH);
  k_feats<<<grid, 128, 0, stream>>>(M1n, selIdx, outFeats);
}
